// CausalAttention_11493332484672
// MI455X (gfx1250) — matrix operand order repaired and checked by reading, not compiled, not run on hardware
//
#include <hip/hip_runtime.h>
#include <hip/hip_bf16.h>

// ---------------------------------------------------------------------------
// CDNA5 (gfx1250) attention (faithful buggy mask) via bf16 WMMA + TDM.
// wave32; WMMA shape 16x16x32 bf16 -> f32; tensor_load_to_lds double-buffered.
// ---------------------------------------------------------------------------

typedef __bf16 bf16_t;
typedef __attribute__((ext_vector_type(16))) __bf16 v16bf;
typedef __attribute__((ext_vector_type(8)))  __bf16 v8bf;
typedef __attribute__((ext_vector_type(8)))  float  v8f;
typedef __attribute__((ext_vector_type(4)))  unsigned int u32x4;
typedef __attribute__((ext_vector_type(8)))  int          i32x8;
typedef __attribute__((ext_vector_type(4)))  int          i32x4;

#define DIM   1024
#define SEQ   2048
#define NH    16
#define DH    64
#define MTOT  4096          // B*S
#define NEGV  (-1.0e9f)

__device__ inline v8f wmma_bf16(v16bf a, v16bf b, v8f c) {
    return __builtin_amdgcn_wmma_f32_16x16x32_bf16(
        false, a, false, b, (short)0, c, false, false);
}

// contiguous 16 bf16 (32B, 32B-aligned by construction)
__device__ inline v16bf ld16(const bf16_t* p) { return *(const v16bf*)p; }

// A-fragment: two contiguous 8-element chunks (16B each)
__device__ inline v16bf ld_a32(const bf16_t* p0, const bf16_t* p1) {
    v8bf lo = *(const v8bf*)p0;
    v8bf hi = *(const v8bf*)p1;
    v16bf r;
#pragma unroll
    for (int i = 0; i < 8; ++i) { r[i] = lo[i]; r[i + 8] = hi[i]; }
    return r;
}

// ---------------------------------------------------------------------------
// TDM: 2-D tile load Global -> LDS.  Descriptor per CDNA5 ISA ch.8 (D#).
//   data_size = 1 (2-byte elements);  X (=dim0) innermost/contiguous.
//   lds_off   = byte offset of the LDS destination (flat addr low 32 bits).
// ---------------------------------------------------------------------------
__device__ inline void tdm_load_2d(unsigned lds_off, const void* gptr,
                                   unsigned tensor_d0, unsigned tensor_d1,
                                   unsigned tile_d0, unsigned tile_d1,
                                   unsigned long long stride0_elems)
{
    unsigned long long ga = (unsigned long long)(uintptr_t)gptr;
    u32x4 g0;
    g0[0] = 1u;                                            // count=1, user, no gather
    g0[1] = lds_off;                                       // lds_addr (bytes)
    g0[2] = (unsigned)(ga & 0xffffffffu);                  // global_addr[31:0]
    g0[3] = (unsigned)((ga >> 32) & 0x1ffffffu)            // global_addr[56:32]
          | (2u << 30);                                    // type = 2 ("image")
    i32x8 g1;
    g1[0] = (int)(1u << 16);                               // wg_mask=0, data_size=1 (2B)
    g1[1] = (int)((tensor_d0 & 0xffffu) << 16);            // tensor_dim0[15:0] @ bits 63:48
    g1[2] = (int)((tensor_d0 >> 16) | ((tensor_d1 & 0xffffu) << 16));
    g1[3] = (int)((tensor_d1 >> 16) | (tile_d0 << 16));    // tile_dim0 @ bits 127:112
    g1[4] = (int)(tile_d1);                                // tile_dim1; tile_dim2=0
    g1[5] = (int)(unsigned)(stride0_elems & 0xffffffffu);  // tensor_dim0_stride[31:0]
    g1[6] = (int)(unsigned)(stride0_elems >> 32);          // stride[47:32]; dim1_stride=0
    g1[7] = 0;
    i32x4 z4 = {0, 0, 0, 0};
    i32x8 z8 = {0, 0, 0, 0, 0, 0, 0, 0};
    __builtin_amdgcn_tensor_load_to_lds(g0, g1, z4, z4, z8, 0);
}

// ---------------------------------------------------------------------------
// Kernel 1: fp32 -> bf16 conversion
// ---------------------------------------------------------------------------
__global__ void cvt_f32_bf16(const float* __restrict__ in, bf16_t* __restrict__ out, int n) {
    int i = blockIdx.x * blockDim.x + threadIdx.x;
    if (i < n) out[i] = (bf16_t)in[i];
}

// ---------------------------------------------------------------------------
// Kernel 2: fused QKV projection.  grid.z: 0=Q, 1=K, 2=V(transposed store).
// C[m,n] = sum_k x[m,k] * W[n,k] + bias[n]
// Block = 256 threads (8 waves); block tile 128x128; wave tile 32x64.
// A/B staged in LDS by the Tensor Data Mover, double-buffered (TENSORcnt).
// ---------------------------------------------------------------------------
__global__ __launch_bounds__(256) void qkv_gemm(
    const bf16_t* __restrict__ xb,
    const bf16_t* __restrict__ wq, const bf16_t* __restrict__ wk, const bf16_t* __restrict__ wv,
    const float*  __restrict__ bq, const float*  __restrict__ bk, const float*  __restrict__ bv,
    bf16_t* __restrict__ Qb, bf16_t* __restrict__ Kb, bf16_t* __restrict__ Vt)
{
    __shared__ __attribute__((aligned(32))) bf16_t As[2][128][32];   // rows m, cols k
    __shared__ __attribute__((aligned(32))) bf16_t Bs[2][128][32];   // rows n, cols k

    const int lane = threadIdx.x & 31;
    const int wave = threadIdx.x >> 5;
    const int h8   = lane >> 4;
    const int ln   = lane & 15;
    const int z    = blockIdx.z;

    const int rowA = blockIdx.y * 128;          // x rows of this block
    const int rowB = blockIdx.x * 128;          // W rows (= output cols)
    const int mrow = (wave >> 1) * 32;          // wave row offset in block tile
    const int wc   = (wave & 1) * 64;           // wave col offset in block tile
    const int m0   = rowA + mrow;
    const int n0   = rowB + wc;

    const bf16_t* W    = (z == 0) ? wq : (z == 1) ? wk : wv;
    const float*  bias = (z == 0) ? bq : (z == 1) ? bk : bv;

    const v8f zv = {0.f, 0.f, 0.f, 0.f, 0.f, 0.f, 0.f, 0.f};
    v8f acc[2][4] = {{zv, zv, zv, zv}, {zv, zv, zv, zv}};

    const unsigned ldsA0 = (unsigned)(uintptr_t)&As[0][0][0];
    const unsigned ldsA1 = (unsigned)(uintptr_t)&As[1][0][0];
    const unsigned ldsB0 = (unsigned)(uintptr_t)&Bs[0][0][0];
    const unsigned ldsB1 = (unsigned)(uintptr_t)&Bs[1][0][0];

    const int nch = DIM / 32;                   // 32 K-chunks
    if (wave == 0) {
        tdm_load_2d(ldsA0, xb + (size_t)rowA * DIM +  0, DIM, MTOT, 32, 128, DIM);
        tdm_load_2d(ldsB0, W  + (size_t)rowB * DIM +  0, DIM, DIM,  32, 128, DIM);
        tdm_load_2d(ldsA1, xb + (size_t)rowA * DIM + 32, DIM, MTOT, 32, 128, DIM);
        tdm_load_2d(ldsB1, W  + (size_t)rowB * DIM + 32, DIM, DIM,  32, 128, DIM);
    }

    for (int ic = 0; ic < nch; ++ic) {
        if (wave == 0) {
            if (ic + 1 < nch) __builtin_amdgcn_s_wait_tensorcnt(2);  // chunk ic landed
            else              __builtin_amdgcn_s_wait_tensorcnt(0);  // last chunk
        }
        __syncthreads();                        // chunk ic resident for all waves

        const int bs = ic & 1;
        v16bf a0 = ld_a32(&As[bs][mrow + ln][h8 * 16],      &As[bs][mrow + ln][h8 * 16 + 8]);
        v16bf a1 = ld_a32(&As[bs][mrow + 16 + ln][h8 * 16], &As[bs][mrow + 16 + ln][h8 * 16 + 8]);
#pragma unroll
        for (int t = 0; t < 4; ++t) {
            v16bf bfr = ld16(&Bs[bs][wc + t * 16 + ln][h8 * 16]);
            acc[0][t] = wmma_bf16(a0, bfr, acc[0][t]);
            acc[1][t] = wmma_bf16(a1, bfr, acc[1][t]);
        }
        __syncthreads();                        // buffer bs free for refill
        if (wave == 0 && ic + 2 < nch) {
            const int kc = (ic + 2) * 32;
            tdm_load_2d(bs ? ldsA1 : ldsA0, xb + (size_t)rowA * DIM + kc, DIM, MTOT, 32, 128, DIM);
            tdm_load_2d(bs ? ldsB1 : ldsB0, W  + (size_t)rowB * DIM + kc, DIM, DIM,  32, 128, DIM);
        }
    }

    // ---- epilogue: bias + bf16 store (Q/K normal, V transposed per head)
#pragma unroll
    for (int ab = 0; ab < 2; ++ab) {
#pragma unroll
        for (int t = 0; t < 4; ++t) {
            const int   n  = n0 + t * 16 + ln;
            const float bb = bias[n];
            if (z < 2) {
                bf16_t* O = (z == 0) ? Qb : Kb;           // [MTOT][DIM] == [B,S,H,dh]
#pragma unroll
                for (int r = 0; r < 8; ++r) {
                    const int m = m0 + ab * 16 + (h8 ? 8 + r : r);
                    O[(size_t)m * DIM + n] = (bf16_t)(acc[ab][t][r] + bb);
                }
            } else {
                const int bidx  = m0 >> 11;
                const int sbase = (m0 & (SEQ - 1)) + ab * 16 + (h8 ? 8 : 0);
                const int hh    = n >> 6;
                const int d     = n & 63;
                v8bf pk;
#pragma unroll
                for (int r = 0; r < 8; ++r) pk[r] = (bf16_t)(acc[ab][t][r] + bb);
                *(v8bf*)(Vt + ((size_t)(bidx * NH + hh) * DH + d) * SEQ + sbase) = pk;
            }
        }
    }
}

// ---------------------------------------------------------------------------
// Kernel 3: flash attention with the faithful buggy mask (keep only kv > q).
// Block = 128 threads (4 waves); block handles 64 q-rows of one (b,h);
// each wave owns 16 q-rows, streams kv in chunks of 32.
// ---------------------------------------------------------------------------
__global__ __launch_bounds__(128) void attn(
    const bf16_t* __restrict__ Qb, const bf16_t* __restrict__ Kb,
    const bf16_t* __restrict__ Vt, float* __restrict__ out)
{
    __shared__ __attribute__((aligned(32))) bf16_t pbuf[4][16][32];

    const int lane = threadIdx.x & 31;
    const int wave = threadIdx.x >> 5;
    const int h8   = lane >> 4;
    const int ln   = lane & 15;
    const int b    = blockIdx.y >> 4;
    const int h    = blockIdx.y & 15;
    const int q0   = blockIdx.x * 64 + wave * 16;

    const bf16_t* qrow = Qb + (((size_t)(b * SEQ + q0 + ln) * NH) + h) * DH;
    const v16bf qa0 = ld_a32(qrow +      h8 * 16, qrow +  8 + h8 * 16);
    const v16bf qa1 = ld_a32(qrow + 32 + h8 * 16, qrow + 40 + h8 * 16);

    const v8f zv = {0.f, 0.f, 0.f, 0.f, 0.f, 0.f, 0.f, 0.f};
    v8f   o[4] = {zv, zv, zv, zv};
    float rm[8], rl[8];
#pragma unroll
    for (int r = 0; r < 8; ++r) { rm[r] = -3.0e38f; rl[r] = 0.f; }

    const bf16_t* Vhead = Vt + (size_t)(b * NH + h) * DH * SEQ;

    for (int kv0 = 0; kv0 < SEQ; kv0 += 32) {
        // prefetch next chunk's K/V cachelines (global_prefetch_b8)
        if (kv0 + 32 < SEQ) {
            __builtin_prefetch(Kb + (((size_t)(b * SEQ + kv0 + 32 + ln) * NH) + h) * DH, 0, 3);
            __builtin_prefetch(Vhead + (size_t)ln * SEQ + kv0 + 32, 0, 3);
        }

        // ---- scores: S = Q * K^T (two 16-col tiles, dh accumulated 2x32)
        v8f s0 = zv, s1 = zv;
        {
            const bf16_t* k0 = Kb + (((size_t)(b * SEQ + kv0 + ln) * NH) + h) * DH + h8 * 16;
            s0 = wmma_bf16(qa0, ld16(k0),      s0);
            s0 = wmma_bf16(qa1, ld16(k0 + 32), s0);
            const bf16_t* k1 = Kb + (((size_t)(b * SEQ + kv0 + 16 + ln) * NH) + h) * DH + h8 * 16;
            s1 = wmma_bf16(qa0, ld16(k1),      s1);
            s1 = wmma_bf16(qa1, ld16(k1 + 32), s1);
        }

        // ---- mask + online softmax
#pragma unroll
        for (int r = 0; r < 8; ++r) {
            const int qg = q0 + (h8 ? 8 + r : r);
            const int c0 = kv0 + ln;
            const int c1 = c0 + 16;
            float v0 = (c0 > qg) ? s0[r] * 0.25f : NEGV;   // /sqrt(H) then mask
            float v1 = (c1 > qg) ? s1[r] * 0.25f : NEGV;

            float tmax = fmaxf(v0, v1);
            tmax = fmaxf(tmax, __shfl_xor(tmax, 1, 32));
            tmax = fmaxf(tmax, __shfl_xor(tmax, 2, 32));
            tmax = fmaxf(tmax, __shfl_xor(tmax, 4, 32));
            tmax = fmaxf(tmax, __shfl_xor(tmax, 8, 32));
            const float mnew = fmaxf(rm[r], tmax);

            const float p0 = __expf(v0 - mnew);
            const float p1 = __expf(v1 - mnew);
            float ps = p0 + p1;
            ps += __shfl_xor(ps, 1, 32);
            ps += __shfl_xor(ps, 2, 32);
            ps += __shfl_xor(ps, 4, 32);
            ps += __shfl_xor(ps, 8, 32);

            const float corr = __expf(rm[r] - mnew);
            rl[r] = rl[r] * corr + ps;
            rm[r] = mnew;
#pragma unroll
            for (int t = 0; t < 4; ++t) o[t][r] *= corr;

            const int row = h8 ? 8 + r : r;
            pbuf[wave][row][ln]      = (bf16_t)p0;
            pbuf[wave][row][16 + ln] = (bf16_t)p1;
        }
        __syncthreads();

        // ---- repack P into A-fragment layout (K = 32 kv positions)
        const v16bf pa = ld_a32(&pbuf[wave][ln][h8 * 16], &pbuf[wave][ln][h8 * 16 + 8]);
        __syncthreads();

        // ---- O += P * V   (B[k,d] = Vt[d][kv0+k], contiguous per lane)
#pragma unroll
        for (int t = 0; t < 4; ++t) {
            const bf16_t* vp = Vhead + (size_t)(t * 16 + ln) * SEQ + kv0 + h8 * 16;
            o[t] = wmma_bf16(pa, ld16(vp), o[t]);
        }
    }

    // ---- final normalization + store [B,S,D] fp32
#pragma unroll
    for (int t = 0; t < 4; ++t) {
#pragma unroll
        for (int r = 0; r < 8; ++r) {
            const int qg = q0 + (h8 ? 8 + r : r);
            out[(size_t)(b * SEQ + qg) * DIM + h * DH + t * 16 + ln] = o[t][r] / rl[r];
        }
    }
}

// ---------------------------------------------------------------------------
// Host launcher
// ---------------------------------------------------------------------------
extern "C" void kernel_launch(void* const* d_in, const int* in_sizes, int n_in,
                              void* d_out, int out_size, void* d_ws, size_t ws_size,
                              hipStream_t stream) {
    const float* x  = (const float*)d_in[0];
    const float* Wq = (const float*)d_in[1];
    const float* bq = (const float*)d_in[2];
    const float* Wk = (const float*)d_in[3];
    const float* bk = (const float*)d_in[4];
    const float* Wv = (const float*)d_in[5];
    const float* bv = (const float*)d_in[6];
    float* out = (float*)d_out;

    char* ws = (char*)d_ws;
    bf16_t* xb  = (bf16_t*)(ws);                         // 8 MiB
    bf16_t* wqb = (bf16_t*)(ws + (8u  << 20));           // 2 MiB each
    bf16_t* wkb = (bf16_t*)(ws + (10u << 20));
    bf16_t* wvb = (bf16_t*)(ws + (12u << 20));
    bf16_t* Qb  = (bf16_t*)(ws + (14u << 20));           // 8 MiB
    bf16_t* Kb  = (bf16_t*)(ws + (22u << 20));           // 8 MiB
    bf16_t* Vt  = (bf16_t*)(ws + (30u << 20));           // 8 MiB (transposed)

    const int NX = MTOT * DIM;
    const int NW = DIM * DIM;
    cvt_f32_bf16<<<(NX + 255) / 256, 256, 0, stream>>>(x,  xb,  NX);
    cvt_f32_bf16<<<(NW + 255) / 256, 256, 0, stream>>>(Wq, wqb, NW);
    cvt_f32_bf16<<<(NW + 255) / 256, 256, 0, stream>>>(Wk, wkb, NW);
    cvt_f32_bf16<<<(NW + 255) / 256, 256, 0, stream>>>(Wv, wvb, NW);

    qkv_gemm<<<dim3(DIM / 128, MTOT / 128, 3), 256, 0, stream>>>(
        xb, wqb, wkb, wvb, bq, bk, bv, Qb, Kb, Vt);

    attn<<<dim3(SEQ / 64, 2 * NH), 128, 0, stream>>>(Qb, Kb, Vt, out);
}
